// MultiheadAttention1_76536317214733
// MI455X (gfx1250) — matrix operand order repaired and checked by reading, not compiled, not run on hardware
//
#include <hip/hip_runtime.h>

// Problem constants (reference: S=2048, B=2, E=1024, H=16, hd=64)
#define S_LEN 2048
#define BSZ   2
#define EMB   1024
#define NH    16
#define HD    64

typedef __bf16 bf16_t;
typedef __attribute__((ext_vector_type(4)))  __bf16 v4bf;
typedef __attribute__((ext_vector_type(8)))  __bf16 v8bf;
typedef __attribute__((ext_vector_type(16))) __bf16 v16bf;
typedef __attribute__((ext_vector_type(4)))  float  v4f;
typedef __attribute__((ext_vector_type(8)))  float  v8f;

__device__ __forceinline__ v8f vzero8() {
  v8f z;
#pragma unroll
  for (int i = 0; i < 8; ++i) z[i] = 0.0f;
  return z;
}

__device__ __forceinline__ v8f wmma_bf16(v16bf a, v16bf b, v8f c) {
  // D = A(16x32 bf16) x B(32x16 bf16) + C(16x16 f32)
  return __builtin_amdgcn_wmma_f32_16x16x32_bf16(false, a, false, b, (short)0, c,
                                                 false, false);
}

// A-fragment loader: lane holds row = lane&15; hi = lane>>4 selects K chunks
// {hi*8 .. hi*8+7} and {hi*8+16 .. hi*8+23} within the 32-wide K slab.
// p must already include row base + chunk base (i.e. ... + hi*8).
__device__ __forceinline__ v16bf load_a16(const bf16_t* p) {
  v8bf lo = *(const v8bf*)p;
  v8bf hi = *(const v8bf*)(p + 16);
  return __builtin_shufflevector(lo, hi, 0, 1, 2, 3, 4, 5, 6, 7, 8, 9, 10, 11,
                                 12, 13, 14, 15);
}

// ---------------------------------------------------------------------------
// fp32 -> bf16 bulk convert (4-wide)
// ---------------------------------------------------------------------------
__global__ __launch_bounds__(256) void k_cvt4(const v4f* __restrict__ x,
                                              v4bf* __restrict__ y, int n4) {
  int i = blockIdx.x * blockDim.x + threadIdx.x;
  if (i < n4) {
    v4f v = x[i];
    v4bf o;
#pragma unroll
    for (int e = 0; e < 4; ++e) o[e] = (bf16_t)v[e];
    y[i] = o;
  }
}

// ---------------------------------------------------------------------------
// In-projection GEMM: X[4096,1024] @ W^T (W=[3072,1024]) + bias.
// Wave tile 64(M) x 64(N): 16 WMMAs per K=32 step. A 64-wide N tile lies
// entirely inside one q/k/v section (64 | 1024), so the scatter target is
// wave-uniform -> readfirstlane + scalar branch.
// ---------------------------------------------------------------------------
__global__ __launch_bounds__(256) void k_inproj(
    const bf16_t* __restrict__ X, const bf16_t* __restrict__ W,
    const float* __restrict__ bias, bf16_t* __restrict__ Qs,
    bf16_t* __restrict__ Kb, bf16_t* __restrict__ Vt) {
  int wave = blockIdx.x * 8 + (threadIdx.x >> 5);
  int lane = threadIdx.x & 31;
  int r = lane & 15, hi = lane >> 4;
  int tM = wave & 63, tN = wave >> 6;  // 64 x 48 wave tiles
  int m0 = tM * 64, n0 = tN * 64;

  v8f acc[4][4];
#pragma unroll
  for (int i = 0; i < 4; ++i)
#pragma unroll
    for (int j = 0; j < 4; ++j) acc[i][j] = vzero8();

  const bf16_t* Xr[4];
#pragma unroll
  for (int i = 0; i < 4; ++i) Xr[i] = X + (size_t)(m0 + i * 16 + r) * EMB;
  const bf16_t* Wr[4];
#pragma unroll
  for (int j = 0; j < 4; ++j) Wr[j] = W + (size_t)(n0 + j * 16 + r) * EMB;

  for (int k = 0; k < EMB; k += 32) {
    v16bf a[4];
#pragma unroll
    for (int i = 0; i < 4; ++i) a[i] = load_a16(Xr[i] + k + hi * 8);
#pragma unroll
    for (int j = 0; j < 4; ++j) {
      v16bf bf = load_a16(Wr[j] + k + hi * 8);
#pragma unroll
      for (int i = 0; i < 4; ++i) acc[i][j] = wmma_bf16(a[i], bf, acc[i][j]);
    }
  }

  // wave-uniform section: 0=q, 1=k, 2=v
  int sec = __builtin_amdgcn_readfirstlane(n0 >> 10);
  int cb = n0 & 1023;  // within-section column base
  if (sec == 0) {
#pragma unroll
    for (int j = 0; j < 4; ++j) {
      int e2 = cb + j * 16 + r;
      int h = e2 >> 6, d = e2 & 63;
      float bv = bias[n0 + j * 16 + r];
#pragma unroll
      for (int i = 0; i < 4; ++i)
#pragma unroll
        for (int e = 0; e < 8; ++e) {
          int row = m0 + i * 16 + e + 8 * hi;  // row = s*B + b
          int b = row & 1, s = row >> 1;
          Qs[((size_t)(b * NH + h) * S_LEN + s) * HD + d] =
              (bf16_t)((acc[i][j][e] + bv) * 0.125f);
        }
    }
  } else if (sec == 1) {
#pragma unroll
    for (int j = 0; j < 4; ++j) {
      int e2 = cb + j * 16 + r;
      int h = e2 >> 6, d = e2 & 63;
      float bv = bias[n0 + j * 16 + r];
#pragma unroll
      for (int i = 0; i < 4; ++i)
#pragma unroll
        for (int e = 0; e < 8; ++e) {
          int row = m0 + i * 16 + e + 8 * hi;
          int b = row & 1, s = row >> 1;
          Kb[((size_t)(b * NH + h) * S_LEN + s) * HD + d] =
              (bf16_t)(acc[i][j][e] + bv);
        }
    }
  } else {
#pragma unroll
    for (int j = 0; j < 4; ++j) {
      int e2 = cb + j * 16 + r;
      int h = e2 >> 6, d = e2 & 63;
      float bv = bias[n0 + j * 16 + r];
#pragma unroll
      for (int i = 0; i < 4; ++i)
#pragma unroll
        for (int e = 0; e < 8; ++e) {
          int row = m0 + i * 16 + e + 8 * hi;
          int b = row & 1, s = row >> 1;
          Vt[((size_t)(b * NH + h) * HD + d) * S_LEN + s] =
              (bf16_t)(acc[i][j][e] + bv);
        }
    }
  }
}

// ---------------------------------------------------------------------------
// Softmax statistics: per (b,h, 16-row q tile) stream all 2048 keys,
// online per-lane max/sumexp, shfl_xor merge across 16 lanes per row.
// ---------------------------------------------------------------------------
__global__ __launch_bounds__(256) void k_stats(const bf16_t* __restrict__ Qs,
                                               const bf16_t* __restrict__ Kb,
                                               float* __restrict__ Ms,
                                               float* __restrict__ Ls) {
  int wave = blockIdx.x * 8 + (threadIdx.x >> 5);
  int lane = threadIdx.x & 31;
  int r = lane & 15, hi = lane >> 4;
  int qt = wave & 127, bh = wave >> 7;  // bh = b*NH + h

  const bf16_t* Qp = Qs + ((size_t)bh * S_LEN + qt * 16 + r) * HD;
  v16bf qa0 = load_a16(Qp + hi * 8);
  v16bf qa1 = load_a16(Qp + 32 + hi * 8);

  float m8[8], l8[8];
#pragma unroll
  for (int e = 0; e < 8; ++e) { m8[e] = -1e30f; l8[e] = 0.0f; }

  for (int t0 = 0; t0 < S_LEN; t0 += 16) {
    const bf16_t* Kp = Kb + ((size_t)bh * S_LEN + t0 + r) * HD;
    v8f acc = vzero8();
    acc = wmma_bf16(qa0, load_a16(Kp + hi * 8), acc);
    acc = wmma_bf16(qa1, load_a16(Kp + 32 + hi * 8), acc);
#pragma unroll
    for (int e = 0; e < 8; ++e) {
      float sv = acc[e];
      float nm = fmaxf(m8[e], sv);
      l8[e] = l8[e] * __expf(m8[e] - nm) + __expf(sv - nm);
      m8[e] = nm;
    }
  }
  // merge the 16 lanes (bit 0..3) holding the same row
#pragma unroll
  for (int mask = 1; mask < 16; mask <<= 1) {
#pragma unroll
    for (int e = 0; e < 8; ++e) {
      float om = __shfl_xor(m8[e], mask, 32);
      float ol = __shfl_xor(l8[e], mask, 32);
      float nm = fmaxf(m8[e], om);
      l8[e] = l8[e] * __expf(m8[e] - nm) + ol * __expf(om - nm);
      m8[e] = nm;
    }
  }
  if (r == 0) {
#pragma unroll
    for (int e = 0; e < 8; ++e) {
      size_t idx = (size_t)bh * S_LEN + qt * 16 + e + 8 * hi;
      Ms[idx] = m8[e];
      Ls[idx] = 1.0f / l8[e];
    }
  }
}

// ---------------------------------------------------------------------------
// P@V with precomputed stats. Score tile -> normalized P (bf16, via LDS
// C->A relayout) -> 4 PV WMMAs accumulating the 16x64 head output tile.
// ---------------------------------------------------------------------------
__global__ __launch_bounds__(256) void k_pv(
    const bf16_t* __restrict__ Qs, const bf16_t* __restrict__ Kb,
    const bf16_t* __restrict__ Vt, const float* __restrict__ Ms,
    const float* __restrict__ Ls, bf16_t* __restrict__ HO) {
  __shared__ __align__(16) bf16_t stage[8][16][40];  // padded 16x32 per wave
  int w = threadIdx.x >> 5;
  int wave = blockIdx.x * 8 + w;
  int lane = threadIdx.x & 31;
  int r = lane & 15, hi = lane >> 4;
  int qt = wave & 127, bh = wave >> 7;

  const bf16_t* Qp = Qs + ((size_t)bh * S_LEN + qt * 16 + r) * HD;
  v16bf qa0 = load_a16(Qp + hi * 8);
  v16bf qa1 = load_a16(Qp + 32 + hi * 8);

  float m8[8], il8[8];
#pragma unroll
  for (int e = 0; e < 8; ++e) {
    size_t idx = (size_t)bh * S_LEN + qt * 16 + e + 8 * hi;
    m8[e] = Ms[idx];
    il8[e] = Ls[idx];
  }

  v8f o[4];
#pragma unroll
  for (int j = 0; j < 4; ++j) o[j] = vzero8();

  for (int t0 = 0; t0 < S_LEN; t0 += 32) {
#pragma unroll
    for (int kt = 0; kt < 2; ++kt) {
      const bf16_t* Kp = Kb + ((size_t)bh * S_LEN + t0 + kt * 16 + r) * HD;
      v8f acc = vzero8();
      acc = wmma_bf16(qa0, load_a16(Kp + hi * 8), acc);
      acc = wmma_bf16(qa1, load_a16(Kp + 32 + hi * 8), acc);
#pragma unroll
      for (int e = 0; e < 8; ++e) {
        stage[w][e + 8 * hi][kt * 16 + r] =
            (bf16_t)(__expf(acc[e] - m8[e]) * il8[e]);
      }
    }
    asm volatile("s_wait_dscnt 0x0" ::: "memory");
    v16bf pa = load_a16(&stage[w][r][hi * 8]);
#pragma unroll
    for (int j = 0; j < 4; ++j) {
      const bf16_t* Vp =
          Vt + ((size_t)bh * HD + j * 16 + r) * S_LEN + t0 + hi * 8;
      o[j] = wmma_bf16(pa, load_a16(Vp), o[j]);
    }
  }

  int b = bh >> 4, h = bh & 15;
#pragma unroll
  for (int j = 0; j < 4; ++j) {
#pragma unroll
    for (int e = 0; e < 8; ++e) {
      int srow = qt * 16 + e + 8 * hi;
      size_t mrow = (size_t)srow * BSZ + b;
      HO[mrow * EMB + h * 64 + j * 16 + r] = (bf16_t)o[j][e];
    }
  }
}

// ---------------------------------------------------------------------------
// avg_w: recompute score tiles (L2-resident K), loop heads per (b,st,tt)
// tile so each output element has exactly one writer (no atomics).
// ---------------------------------------------------------------------------
__global__ __launch_bounds__(256) void k_avg(const bf16_t* __restrict__ Qs,
                                             const bf16_t* __restrict__ Kb,
                                             const float* __restrict__ Ms,
                                             const float* __restrict__ Ls,
                                             float* __restrict__ avg) {
  int wave = blockIdx.x * 8 + (threadIdx.x >> 5);
  int lane = threadIdx.x & 31;
  int r = lane & 15, hi = lane >> 4;
  int tt = wave & 127;
  int st = (wave >> 7) & 127;
  int b = wave >> 14;

  float accm[8];
#pragma unroll
  for (int e = 0; e < 8; ++e) accm[e] = 0.0f;

  for (int h = 0; h < NH; ++h) {
    int bh = b * NH + h;
    const bf16_t* Qp = Qs + ((size_t)bh * S_LEN + st * 16 + r) * HD;
    v16bf qa0 = load_a16(Qp + hi * 8);
    v16bf qa1 = load_a16(Qp + 32 + hi * 8);
    const bf16_t* Kp = Kb + ((size_t)bh * S_LEN + tt * 16 + r) * HD;
    v8f acc = vzero8();
    acc = wmma_bf16(qa0, load_a16(Kp + hi * 8), acc);
    acc = wmma_bf16(qa1, load_a16(Kp + 32 + hi * 8), acc);
#pragma unroll
    for (int e = 0; e < 8; ++e) {
      size_t sidx = (size_t)bh * S_LEN + st * 16 + e + 8 * hi;
      accm[e] += __expf(acc[e] - Ms[sidx]) * Ls[sidx];
    }
  }
#pragma unroll
  for (int e = 0; e < 8; ++e) {
    int srow = st * 16 + e + 8 * hi;
    avg[((size_t)b * S_LEN + srow) * S_LEN + tt * 16 + r] =
        accm[e] * (1.0f / NH);
  }
}

// ---------------------------------------------------------------------------
// Out-projection GEMM: HO[4096,1024] @ Wo^T + bias -> attn_out (fp32)
// Wave tile 64(M) x 64(N).
// ---------------------------------------------------------------------------
__global__ __launch_bounds__(256) void k_outproj(const bf16_t* __restrict__ A,
                                                 const bf16_t* __restrict__ W,
                                                 const float* __restrict__ bias,
                                                 float* __restrict__ out) {
  int wave = blockIdx.x * 8 + (threadIdx.x >> 5);
  int lane = threadIdx.x & 31;
  int r = lane & 15, hi = lane >> 4;
  int tM = wave & 63, tN = wave >> 6;  // 64 x 16 wave tiles
  int m0 = tM * 64, n0 = tN * 64;

  v8f acc[4][4];
#pragma unroll
  for (int i = 0; i < 4; ++i)
#pragma unroll
    for (int j = 0; j < 4; ++j) acc[i][j] = vzero8();

  const bf16_t* Ar[4];
#pragma unroll
  for (int i = 0; i < 4; ++i) Ar[i] = A + (size_t)(m0 + i * 16 + r) * EMB;
  const bf16_t* Wr[4];
#pragma unroll
  for (int j = 0; j < 4; ++j) Wr[j] = W + (size_t)(n0 + j * 16 + r) * EMB;

  for (int k = 0; k < EMB; k += 32) {
    v16bf a[4];
#pragma unroll
    for (int i = 0; i < 4; ++i) a[i] = load_a16(Ar[i] + k + hi * 8);
#pragma unroll
    for (int j = 0; j < 4; ++j) {
      v16bf bf = load_a16(Wr[j] + k + hi * 8);
#pragma unroll
      for (int i = 0; i < 4; ++i) acc[i][j] = wmma_bf16(a[i], bf, acc[i][j]);
    }
  }

#pragma unroll
  for (int j = 0; j < 4; ++j) {
    int c = n0 + j * 16 + r;
    float bv = bias[c];
#pragma unroll
    for (int i = 0; i < 4; ++i)
#pragma unroll
      for (int e = 0; e < 8; ++e) {
        int row = m0 + i * 16 + e + 8 * hi;
        out[(size_t)row * EMB + c] = acc[i][j][e] + bv;
      }
  }
}

// ---------------------------------------------------------------------------
extern "C" void kernel_launch(void* const* d_in, const int* in_sizes, int n_in,
                              void* d_out, int out_size, void* d_ws,
                              size_t ws_size, hipStream_t stream) {
  const float* X = (const float*)d_in[0];    // query [S,B,E]
  const float* Wi = (const float*)d_in[1];   // in_proj_weight [3E,E]
  const float* bi = (const float*)d_in[2];   // in_proj_bias [3E]
  const float* Wo = (const float*)d_in[3];   // out_proj_weight [E,E]
  const float* bo = (const float*)d_in[4];   // out_proj_bias [E]

  char* ws = (char*)d_ws;
  bf16_t* Xbf  = (bf16_t*)(ws + 0);          //  8,388,608 B
  bf16_t* Wbf  = (bf16_t*)(ws + 8388608);    //  6,291,456 B
  bf16_t* Wobf = (bf16_t*)(ws + 14680064);   //  2,097,152 B
  bf16_t* Qs   = (bf16_t*)(ws + 16777216);   //  8,388,608 B  [b*16+h][s][d]
  bf16_t* Kb   = (bf16_t*)(ws + 25165824);   //  8,388,608 B  [b*16+h][t][d]
  bf16_t* Vt   = (bf16_t*)(ws + 33554432);   //  8,388,608 B  [b*16+h][d][t]
  bf16_t* HO   = (bf16_t*)(ws + 41943040);   //  8,388,608 B  [s*B+b][E]
  float*  Ms   = (float*)(ws + 50331648);    //    262,144 B
  float*  Ls   = (float*)(ws + 50593792);    //    262,144 B

  float* attn_out = (float*)d_out;                        // S*B*E floats
  float* avg = attn_out + (size_t)S_LEN * BSZ * EMB;      // B*S*S floats

  // 0) one-time fp32 -> bf16 conversions
  k_cvt4<<<4096, 256, 0, stream>>>((const v4f*)X, (v4bf*)Xbf, 1048576);
  k_cvt4<<<3072, 256, 0, stream>>>((const v4f*)Wi, (v4bf*)Wbf, 786432);
  k_cvt4<<<1024, 256, 0, stream>>>((const v4f*)Wo, (v4bf*)Wobf, 262144);

  // 1) packed QKV projection -> Qs (scaled), Kb, Vt  (64x64 wave tiles)
  k_inproj<<<384, 256, 0, stream>>>(Xbf, Wbf, bi, Qs, Kb, Vt);

  // 2) softmax row statistics (max, 1/sumexp)
  k_stats<<<512, 256, 0, stream>>>(Qs, Kb, Ms, Ls);

  // 3) normalized P @ V -> head outputs (bf16)
  k_pv<<<512, 256, 0, stream>>>(Qs, Kb, Vt, Ms, Ls, HO);

  // 4) avg attention weights (recompute scores, loop heads, exclusive writes)
  k_avg<<<4096, 256, 0, stream>>>(Qs, Kb, Ms, Ls, avg);

  // 5) output projection -> attn_out (fp32)  (64x64 wave tiles)
  k_outproj<<<128, 256, 0, stream>>>(HO, Wobf, bo, attn_out);
}
